// AttentionHead_58987080843316
// MI455X (gfx1250) — matrix operand order repaired and checked by reading, not compiled, not run on hardware
//
#include <hip/hip_runtime.h>
#include <stdint.h>

// ---------- types ----------
typedef __attribute__((ext_vector_type(16))) __bf16 v16bf;
typedef __attribute__((ext_vector_type(8)))  __bf16 v8bf;
typedef __attribute__((ext_vector_type(8)))  float  v8f;

static __device__ __forceinline__ unsigned short f32_to_bf16(float f) {
  union { __bf16 h; unsigned short s; } t;
  t.h = (__bf16)f;
  return t.s;
}
static __device__ __forceinline__ unsigned pack2(float a, float b) {
  union { __bf16 h[2]; unsigned u; } t;
  t.h[0] = (__bf16)a; t.h[1] = (__bf16)b;
  return t.u;
}
static __device__ __forceinline__ v16bf cat8(v8bf lo, v8bf hi) {
  return __builtin_shufflevector(lo, hi, 0,1,2,3,4,5,6,7,8,9,10,11,12,13,14,15);
}
static __device__ __forceinline__ v8f wmma_bf16(v16bf a, v16bf b, v8f c) {
  return __builtin_amdgcn_wmma_f32_16x16x32_bf16(false, a, false, b, (short)0, c, false, false);
}
union AFU { v16bf v; unsigned u[8]; };

// ---------- problem constants ----------
#define BATCH 4
#define SEQ   4096
#define DMODEL 1024
#define KDIM  128
#define NROWS (BATCH*SEQ)          // 16384
#define NTILES (NROWS/16)          // 1024 row tiles
// workspace layout
#define WP_BYTES  (3u*256u*32u*32u)              // 786432: 3 W, 256 frags, 32 lanes, 32B
#define QK_BYTES  ((size_t)NROWS*KDIM*2)         // 4 MB each

// =====================================================================
// Kernel 1: pack W [1024x128] fp32 -> bf16 B-fragments.
// Fragment (kt 0..31, nt 0..7); per lane 16 halves (32B) contiguous:
// lane l: n = nt*16 + (l&15); halves j pair = W[kt*32 + (l>>4)*16 + 2j (+1)][n]
// =====================================================================
__global__ __launch_bounds__(256) void pack_w_kernel(
    const float* __restrict__ wq, const float* __restrict__ wk,
    const float* __restrict__ wv, unsigned short* __restrict__ Wp)
{
  int tid = blockIdx.x * blockDim.x + threadIdx.x;
  if (tid >= 3 * 256 * 32) return;
  int l  = tid & 31;
  int fi = (tid >> 5) & 255;
  int w  = tid >> 13;
  int kt = fi >> 3, nt = fi & 7;
  const float* W = (w == 0) ? wq : (w == 1) ? wk : wv;
  int n  = nt * 16 + (l & 15);
  int kh = l >> 4;
  unsigned o[8];
#pragma unroll
  for (int j = 0; j < 8; ++j) {
    int d0 = kt * 32 + kh * 16 + 2 * j;
    o[j] = pack2(W[(size_t)d0 * KDIM + n], W[(size_t)(d0 + 1) * KDIM + n]);
  }
  uint4* dst = (uint4*)(Wp + (size_t)tid * 16);
  dst[0] = make_uint4(o[0], o[1], o[2], o[3]);
  dst[1] = make_uint4(o[4], o[5], o[6], o[7]);
}

// =====================================================================
// Kernel 2: QKV projection. One wave per 16-row tile of X.
// kt-outer: A fragment converted once, shared by all 3 weights (acc[3][8]).
// Q,K row-major bf16 [B,S,128]; V transposed bf16 [B,128,S].
// =====================================================================
__global__ __launch_bounds__(256) void qkv_kernel(
    const float* __restrict__ X, const unsigned short* __restrict__ Wp,
    unsigned short* __restrict__ Qp, unsigned short* __restrict__ Kp,
    unsigned short* __restrict__ Vt)
{
  int lane = threadIdx.x & 31;
  int wave = blockIdx.x * (blockDim.x >> 5) + (threadIdx.x >> 5);
  if (wave >= NTILES) return;
  int rowBase = wave * 16;
  int h  = lane >> 4;
  int ln = lane & 15;
  const float* xrow = X + (size_t)(rowBase + ln) * DMODEL;
  int batch = rowBase >> 12;
  int sBase = rowBase & (SEQ - 1);

  v8f acc[3][8];
#pragma unroll
  for (int w = 0; w < 3; ++w)
#pragma unroll
    for (int nt = 0; nt < 8; ++nt) acc[w][nt] = (v8f)0.0f;

  for (int kt = 0; kt < 32; ++kt) {
    int c0 = kt * 32 + h * 16;                  // halves 0..7 -> K c0..c0+7
    const float4* p0 = (const float4*)(xrow + c0);
    float4 f0 = p0[0], f1 = p0[1];
    const float4* p1 = (const float4*)(xrow + c0 + 8); // halves 8..15 -> K c0+8..+15
    float4 f2 = p1[0], f3 = p1[1];
    if (kt + 1 < 32) __builtin_prefetch(xrow + c0 + 32, 0, 1);
    AFU t;
    t.u[0] = pack2(f0.x, f0.y); t.u[1] = pack2(f0.z, f0.w);
    t.u[2] = pack2(f1.x, f1.y); t.u[3] = pack2(f1.z, f1.w);
    t.u[4] = pack2(f2.x, f2.y); t.u[5] = pack2(f2.z, f2.w);
    t.u[6] = pack2(f3.x, f3.y); t.u[7] = pack2(f3.z, f3.w);
    v16bf a = t.v;
#pragma unroll
    for (int w = 0; w < 3; ++w) {
      v16bf b[8];
#pragma unroll
      for (int nt = 0; nt < 8; ++nt) {
        const v8bf* bp = (const v8bf*)(Wp +
            (((size_t)w * 256 + (size_t)kt * 8 + nt) * 32 + lane) * 16);
        b[nt] = cat8(bp[0], bp[1]);
      }
#pragma unroll
      for (int nt = 0; nt < 8; ++nt)
        acc[w][nt] = wmma_bf16(a, b[nt], acc[w][nt]);
    }
  }

  // D layout: lane column n = nt*16+ln ; rows m = r + 8*h
#pragma unroll
  for (int w = 0; w < 2; ++w) {
    unsigned short* dst = (w == 0 ? Qp : Kp) + (size_t)rowBase * KDIM;
#pragma unroll
    for (int nt = 0; nt < 8; ++nt) {
      int n = nt * 16 + ln;
#pragma unroll
      for (int r = 0; r < 8; ++r)
        dst[(size_t)(r + 8 * h) * KDIM + n] = f32_to_bf16(acc[w][nt][r]);
    }
  }
#pragma unroll
  for (int nt = 0; nt < 8; ++nt) {
    int n = nt * 16 + ln;
    unsigned short* dst = Vt + (size_t)batch * KDIM * SEQ +
                          (size_t)n * SEQ + sBase + 8 * h;
    uint4 pk = make_uint4(pack2(acc[2][nt][0], acc[2][nt][1]),
                          pack2(acc[2][nt][2], acc[2][nt][3]),
                          pack2(acc[2][nt][4], acc[2][nt][5]),
                          pack2(acc[2][nt][6], acc[2][nt][7]));
    *(uint4*)dst = pk;                          // 8 consecutive keys, 16B aligned
  }
}

// =====================================================================
// Kernel 3: flash attention. One wave per 16-query tile; 128 key blocks x 32.
// K and V fragments are preloaded at the top of each iteration so their
// latency overlaps the QK WMMAs and softmax VALU work. Softmax runs in
// base-2 (logits pre-scaled by log2(e)/sqrt(128)) so each exponential is a
// single v_exp_f32.
// =====================================================================
__global__ __launch_bounds__(256) void flash_kernel(
    const unsigned short* __restrict__ Qp, const unsigned short* __restrict__ Kp,
    const unsigned short* __restrict__ Vt, float* __restrict__ out)
{
  __shared__ __align__(16) unsigned short ldsP[8][16][32];   // per-wave 16x32 P tile
  int lane = threadIdx.x & 31;
  int wv   = threadIdx.x >> 5;
  int wave = blockIdx.x * 8 + wv;
  if (wave >= NTILES) return;
  int batch = wave >> 8;                 // 256 tiles per batch
  int qBase = (wave & 255) * 16;
  int h  = lane >> 4;
  int ln = lane & 15;

  // resident Q A-fragments (16 rows x 128 features)
  const unsigned short* Qrow = Qp + ((size_t)batch * SEQ + qBase + ln) * KDIM;
  v16bf aQ[4];
#pragma unroll
  for (int kt = 0; kt < 4; ++kt) {
    const v8bf* plo = (const v8bf*)(Qrow + kt * 32 + h * 16);
    const v8bf* phi = (const v8bf*)(Qrow + kt * 32 + h * 16 + 8);
    aQ[kt] = cat8(plo[0], phi[0]);
  }

  float mI[8], lI[8];
  v8f acc[8];
#pragma unroll
  for (int r = 0; r < 8; ++r) { mI[r] = -1e30f; lI[r] = 0.0f; }
#pragma unroll
  for (int nt = 0; nt < 8; ++nt) acc[nt] = (v8f)0.0f;

  const unsigned short* Kb = Kp + (size_t)batch * SEQ * KDIM;
  const unsigned short* Vb = Vt + (size_t)batch * KDIM * SEQ;
  // 1/sqrt(128) * log2(e): softmax done in base 2 (identical result)
  const float scl = (float)(0.08838834764831845 * 1.4426950408889634);

  for (int kb = 0; kb < 128; ++kb) {
    int keyBase = kb * 32;

    // ---- preload K fragments (2 key halves x 4 k-steps) ----
    v16bf kf[8];
#pragma unroll
    for (int kv = 0; kv < 2; ++kv) {
      const unsigned short* krow =
          Kb + (size_t)(keyBase + kv * 16 + ln) * KDIM + h * 16;
#pragma unroll
      for (int kt = 0; kt < 4; ++kt) {
        const v8bf* p = (const v8bf*)(krow + kt * 32);
        kf[kv * 4 + kt] = cat8(p[0], p[1]);
      }
    }
    // ---- preload V fragments (independent of softmax) ----
    v16bf vf[8];
#pragma unroll
    for (int nt = 0; nt < 8; ++nt) {
      const v8bf* p = (const v8bf*)(Vb + (size_t)(nt * 16 + ln) * SEQ +
                                    keyBase + h * 16);
      vf[nt] = cat8(p[0], p[1]);
    }
    if (kb + 1 < 128)
      __builtin_prefetch(Kb + (size_t)(keyBase + 32 + ln) * KDIM, 0, 1);

    // ---- QK^T: two 16x16 score tiles ----
    v8f s0 = (v8f)0.0f, s1 = (v8f)0.0f;
#pragma unroll
    for (int kt = 0; kt < 4; ++kt) s0 = wmma_bf16(aQ[kt], kf[kt], s0);
#pragma unroll
    for (int kt = 0; kt < 4; ++kt) s1 = wmma_bf16(aQ[kt], kf[4 + kt], s1);

    // ---- online softmax over this 16x32 tile (rows owned: r + 8*h) ----
    float p0[8], p1[8], rmax[8], rsum[8], corr[8];
#pragma unroll
    for (int r = 0; r < 8; ++r) {
      s0[r] *= scl; s1[r] *= scl;
      rmax[r] = fmaxf(s0[r], s1[r]);
    }
#pragma unroll
    for (int off = 1; off <= 8; off <<= 1)
#pragma unroll
      for (int r = 0; r < 8; ++r)
        rmax[r] = fmaxf(rmax[r], __shfl_xor(rmax[r], off, 32));
#pragma unroll
    for (int r = 0; r < 8; ++r) {
      float mn = fmaxf(mI[r], rmax[r]);
      corr[r] = exp2f(mI[r] - mn);
      p0[r]   = exp2f(s0[r] - mn);
      p1[r]   = exp2f(s1[r] - mn);
      mI[r]   = mn;
      rsum[r] = p0[r] + p1[r];
    }
#pragma unroll
    for (int off = 1; off <= 8; off <<= 1)
#pragma unroll
      for (int r = 0; r < 8; ++r)
        rsum[r] += __shfl_xor(rsum[r], off, 32);
#pragma unroll
    for (int r = 0; r < 8; ++r) lI[r] = lI[r] * corr[r] + rsum[r];
#pragma unroll
    for (int nt = 0; nt < 8; ++nt)
#pragma unroll
      for (int r = 0; r < 8; ++r) acc[nt][r] *= corr[r];

    // ---- stage P (D layout -> LDS row-major) then reload as A fragment ----
#pragma unroll
    for (int r = 0; r < 8; ++r) {
      int row = r + 8 * h;
      ldsP[wv][row][ln]      = f32_to_bf16(p0[r]);
      ldsP[wv][row][ln + 16] = f32_to_bf16(p1[r]);
    }
    asm volatile("s_wait_dscnt 0x0" ::: "memory");
    const v8bf* pl = (const v8bf*)&ldsP[wv][ln][h * 16];
    const v8bf* ph = (const v8bf*)&ldsP[wv][ln][h * 16 + 8];
    v16bf aP = cat8(pl[0], ph[0]);

    // ---- PV: accumulate 16x128 output (V already in registers) ----
#pragma unroll
    for (int nt = 0; nt < 8; ++nt)
      acc[nt] = wmma_bf16(aP, vf[nt], acc[nt]);
  }

  // epilogue: normalize and store fp32
  float* orow = out + ((size_t)batch * SEQ + qBase) * KDIM;
#pragma unroll
  for (int r = 0; r < 8; ++r) {
    float inv = 1.0f / lI[r];
    size_t rowOff = (size_t)(r + 8 * h) * KDIM;
#pragma unroll
    for (int nt = 0; nt < 8; ++nt)
      orow[rowOff + nt * 16 + ln] = acc[nt][r] * inv;
  }
}

// =====================================================================
extern "C" void kernel_launch(void* const* d_in, const int* in_sizes, int n_in,
                              void* d_out, int out_size, void* d_ws, size_t ws_size,
                              hipStream_t stream) {
  (void)in_sizes; (void)n_in; (void)out_size; (void)ws_size;
  const float* X  = (const float*)d_in[0];
  const float* wq = (const float*)d_in[1];
  const float* wk = (const float*)d_in[2];
  const float* wv = (const float*)d_in[3];
  float* out = (float*)d_out;

  char* ws = (char*)d_ws;
  unsigned short* Wp = (unsigned short*)ws;
  unsigned short* Qp = (unsigned short*)(ws + WP_BYTES);
  unsigned short* Kp = (unsigned short*)(ws + WP_BYTES + QK_BYTES);
  unsigned short* Vt = (unsigned short*)(ws + WP_BYTES + 2 * QK_BYTES);

  pack_w_kernel<<<(3 * 256 * 32 + 255) / 256, 256, 0, stream>>>(wq, wk, wv, Wp);
  qkv_kernel  <<<NTILES / 8, 256, 0, stream>>>(X, Wp, Qp, Kp, Vt);
  flash_kernel<<<NTILES / 8, 256, 0, stream>>>(Qp, Kp, Vt, out);
}
